// MultiHeadedAttention_85830626443478
// MI455X (gfx1250) — matrix operand order repaired and checked by reading, not compiled, not run on hardware
//
#include <hip/hip_runtime.h>

// ---------------------------------------------------------------------------
// Types matching gfx1250 WMMA builtin signatures
// ---------------------------------------------------------------------------
typedef __attribute__((ext_vector_type(16))) __bf16          v16bf;
typedef __attribute__((ext_vector_type(8)))  float           v8f;
typedef __attribute__((ext_vector_type(8)))  unsigned short  us8;

union FragCvt { us8 u[2]; v16bf v; };

static __device__ __forceinline__ unsigned short f2bf(float f) {
    unsigned u = __float_as_uint(f);
    u += 0x7FFFu + ((u >> 16) & 1u);           // round-to-nearest-even
    return (unsigned short)(u >> 16);
}

static __device__ __forceinline__ us8 ld8(const unsigned short* p) {
    return *(const us8*)p;                     // 16B load -> global_load_b128 / ds_load_b128
}

static __device__ __forceinline__ v16bf mkfrag(us8 lo, us8 hi) {
    FragCvt f; f.u[0] = lo; f.u[1] = hi; return f.v;
}

static __device__ __forceinline__ v8f wmma_bf16(v16bf a, v16bf b, v8f c) {
    // v_wmma_f32_16x16x32_bf16
    return __builtin_amdgcn_wmma_f32_16x16x32_bf16(
        /*neg_a=*/false, a, /*neg_b=*/false, b,
        /*c_mod=*/(short)0, c, /*reuse_a=*/false, /*reuse_b=*/false);
}

// ---------------------------------------------------------------------------
// Problem constants
// ---------------------------------------------------------------------------
#define BATCH   4
#define SEQ     1024
#define HEADS   16
#define DHID    64
#define DMODEL  1024      // == HEADS*DHID == D_IN

// ---------------------------------------------------------------------------
// Elementwise conversion / transpose kernels (bf16 staging)
// ---------------------------------------------------------------------------
__global__ __launch_bounds__(256) void cvt_f32_bf16(const float* __restrict__ in,
                                                    unsigned short* __restrict__ out, int n) {
    int i = blockIdx.x * 256 + threadIdx.x;
    if (i < n) out[i] = f2bf(in[i]);
}

// Wt[n*1024 + k] = bf16(W[k*1024 + n]); 1024x1024 weights
__global__ __launch_bounds__(256) void cvt_w_transpose(const float* __restrict__ in,
                                                       unsigned short* __restrict__ out) {
    int i = blockIdx.x * 256 + threadIdx.x;   // i = n*1024 + k
    int n = i >> 10, k = i & 1023;
    out[i] = f2bf(in[k * 1024 + n]);
}

// v_t[((b*H + h)*64 + d)*1024 + s] = v_bf[(b*S + s)*1024 + h*64 + d]
__global__ __launch_bounds__(256) void transpose_v(const unsigned short* __restrict__ vb,
                                                   unsigned short* __restrict__ vt) {
    int i = blockIdx.x * 256 + threadIdx.x;   // index into vt, 4M elements
    int s = i & 1023;
    int t = i >> 10;
    int d = t & 63;  t >>= 6;
    int h = t & 15;  int b = t >> 4;
    vt[i] = vb[(b * SEQ + s) * DMODEL + h * DHID + d];
}

// ---------------------------------------------------------------------------
// GEMM: C[M x N] = A[M x K] * W[K x N] + bias,  K = 1024.
// A in bf16 row-major, W provided TRANSPOSED bf16 (Wt[n*K + k]).
// One wave computes a 16x64 tile (4 accumulators of 16x16).
// ---------------------------------------------------------------------------
template<int BF16OUT>
__global__ __launch_bounds__(256) void gemm_bf16_wmma(const unsigned short* __restrict__ A,
                                                      const unsigned short* __restrict__ Wt,
                                                      const float* __restrict__ bias,
                                                      float* __restrict__ Cf,
                                                      unsigned short* __restrict__ Cb,
                                                      int M, int N) {
    const int K = 1024;
    int wave = blockIdx.x * 8 + (threadIdx.x >> 5);
    int lane = threadIdx.x & 31;
    int li = lane & 15, half = lane >> 4;
    int ntiles = N >> 6;
    int mt = wave / ntiles, nt = wave % ntiles;
    int m0 = mt * 16, n0 = nt * 64;
    if (m0 >= M) return;

    const v8f vz = {0.f, 0.f, 0.f, 0.f, 0.f, 0.f, 0.f, 0.f};
    v8f acc[4] = {vz, vz, vz, vz};

    const unsigned short* abase = A + (size_t)(m0 + li) * K;
    const unsigned short* wbase[4];
    int ncol[4];
#pragma unroll
    for (int j = 0; j < 4; ++j) {
        ncol[j]  = n0 + j * 16 + li;
        wbase[j] = Wt + (size_t)ncol[j] * K;
    }

    for (int kk = 0; kk < K; kk += 32) {
        // A fragment 16x32: K(e) = half*16 + e
        v16bf a = mkfrag(ld8(abase + kk + half * 16),
                         ld8(abase + kk + half * 16 + 8));
#pragma unroll
        for (int j = 0; j < 4; ++j) {
            // B fragment 32x16: K(e) = half*16 + e (contiguous in Wt row)
            const unsigned short* p = wbase[j] + kk + half * 16;
            v16bf b = mkfrag(ld8(p), ld8(p + 8));
            acc[j] = wmma_bf16(a, b, acc[j]);
        }
    }

#pragma unroll
    for (int j = 0; j < 4; ++j) {
        float bv = bias[ncol[j]];
#pragma unroll
        for (int r = 0; r < 8; ++r) {
            int m = m0 + r + 8 * half;       // C layout: VGPR r -> M = r + 8*half
            float v = acc[j][r] + bv;
            if (BF16OUT) Cb[(size_t)m * N + ncol[j]] = f2bf(v);
            else         Cf[(size_t)m * N + ncol[j]] = v;
        }
    }
}

// ---------------------------------------------------------------------------
// Flash attention: one wave handles 16 query rows of one (b,h).
// q_bf/k_bf: [B,S,H*64] bf16 ; v_t: [B,H,64,S] bf16
// outH: [B,H,S,64] fp32 ; attn_bf: [B,S,H*64] bf16 (for output projection)
// ---------------------------------------------------------------------------
__global__ __launch_bounds__(256) void attn_wmma(const unsigned short* __restrict__ qbf,
                                                 const unsigned short* __restrict__ kbf,
                                                 const unsigned short* __restrict__ vt,
                                                 float* __restrict__ outH,
                                                 unsigned short* __restrict__ attn_bf) {
    __shared__ unsigned short p_lds[8 * 512];   // per-wave 16x32 bf16 P tile
    int w    = threadIdx.x >> 5;
    int lane = threadIdx.x & 31;
    int li = lane & 15, half = lane >> 4;

    int wave = blockIdx.x * 8 + w;              // 0..4095
    int qt = wave & 63;
    int h  = (wave >> 6) & 15;
    int b  = wave >> 10;
    int m0 = qt * 16;

    // Q fragments for d=0..31 and d=32..63 (A-layout, held for the whole loop)
    const unsigned short* qbase = qbf + (size_t)(b * SEQ + m0 + li) * DMODEL + h * DHID;
    v16bf aq0 = mkfrag(ld8(qbase + half * 16),      ld8(qbase + half * 16 + 8));
    v16bf aq1 = mkfrag(ld8(qbase + 32 + half * 16), ld8(qbase + 32 + half * 16 + 8));

    const v8f vz = {0.f, 0.f, 0.f, 0.f, 0.f, 0.f, 0.f, 0.f};
    v8f o[4] = {vz, vz, vz, vz};
    float m_i[8], l_i[8];
#pragma unroll
    for (int r = 0; r < 8; ++r) { m_i[r] = -3.0e38f; l_i[r] = 0.f; }

    unsigned short* pl = p_lds + w * 512;
    const unsigned short* vthead = vt + ((size_t)(b * HEADS + h) * DHID) * SEQ;
    const float scale = 0.125f;                 // 1/sqrt(64)

    for (int kb = 0; kb < SEQ; kb += 32) {
        // ----- scores for 32 keys (two 16x16 WMMA tiles) -----
        v8f st[2];
#pragma unroll
        for (int tt = 0; tt < 2; ++tt) {
            const unsigned short* kbase =
                kbf + (size_t)(b * SEQ + kb + tt * 16 + li) * DMODEL + h * DHID;
            // B = K^T: K-dim of WMMA = d; contiguous along K row
            v16bf b0 = mkfrag(ld8(kbase + half * 16),      ld8(kbase + half * 16 + 8));
            v16bf b1 = mkfrag(ld8(kbase + 32 + half * 16), ld8(kbase + 32 + half * 16 + 8));
            v8f s = vz;
            s = wmma_bf16(aq0, b0, s);
            s = wmma_bf16(aq1, b1, s);
            st[tt] = s;
        }

        // ----- online softmax (row reductions across the 16-lane half) -----
        float p0[8], p1[8];
#pragma unroll
        for (int r = 0; r < 8; ++r) {
            float s0 = st[0][r] * scale, s1 = st[1][r] * scale;
            float tmax = fmaxf(s0, s1);
#pragma unroll
            for (int msk = 1; msk < 16; msk <<= 1)
                tmax = fmaxf(tmax, __shfl_xor(tmax, msk, 32));
            float nm   = fmaxf(m_i[r], tmax);
            float corr = __expf(m_i[r] - nm);
            float e0 = __expf(s0 - nm), e1 = __expf(s1 - nm);
            float rs = e0 + e1;
#pragma unroll
            for (int msk = 1; msk < 16; msk <<= 1)
                rs += __shfl_xor(rs, msk, 32);
            l_i[r] = l_i[r] * corr + rs;
            m_i[r] = nm;
            p0[r] = e0; p1[r] = e1;
#pragma unroll
            for (int j = 0; j < 4; ++j) o[j][r] *= corr;
        }

        // ----- C-layout -> A-layout for P via per-wave LDS region -----
#pragma unroll
        for (int r = 0; r < 8; ++r) {
            pl[(r + 8 * half) * 32 + li]      = f2bf(p0[r]);
            pl[(r + 8 * half) * 32 + 16 + li] = f2bf(p1[r]);
        }
        v16bf pa = mkfrag(ld8(pl + li * 32 + half * 16),
                          ld8(pl + li * 32 + half * 16 + 8));

        // ----- O += P(16x32) x V(32x64) -----
#pragma unroll
        for (int j = 0; j < 4; ++j) {
            const unsigned short* vb = vthead + (size_t)(j * 16 + li) * SEQ + kb + half * 16;
            v16bf vf = mkfrag(ld8(vb), ld8(vb + 8));
            o[j] = wmma_bf16(pa, vf, o[j]);
        }
    }

    // ----- finalize and write out_h (fp32) + merged bf16 for Wo GEMM -----
#pragma unroll
    for (int r = 0; r < 8; ++r) {
        float inv = 1.0f / l_i[r];
        int m = m0 + r + 8 * half;
#pragma unroll
        for (int j = 0; j < 4; ++j) {
            int d = j * 16 + li;
            float v = o[j][r] * inv;
            outH[(((size_t)(b * HEADS + h) * SEQ) + m) * DHID + d] = v;
            attn_bf[(size_t)(b * SEQ + m) * DMODEL + h * DHID + d] = f2bf(v);
        }
    }
}

// ---------------------------------------------------------------------------
// Host-side launch sequence (all on `stream`, workspace-only scratch)
// ---------------------------------------------------------------------------
extern "C" void kernel_launch(void* const* d_in, const int* in_sizes, int n_in,
                              void* d_out, int out_size, void* d_ws, size_t ws_size,
                              hipStream_t stream) {
    (void)in_sizes; (void)n_in; (void)out_size; (void)ws_size;

    const float* query = (const float*)d_in[0];
    // d_in[1] ("keys") intentionally unused: reference computes k from query.
    const float* value = (const float*)d_in[2];
    const float* Wq = (const float*)d_in[3];
    const float* bq = (const float*)d_in[4];
    const float* Wk = (const float*)d_in[5];
    const float* bk = (const float*)d_in[6];
    const float* Wv = (const float*)d_in[7];
    const float* bv = (const float*)d_in[8];
    const float* Wo = (const float*)d_in[9];
    const float* bo = (const float*)d_in[10];

    float* out_proj = (float*)d_out;                       // [B,S,1024] fp32
    float* out_h    = (float*)d_out + (size_t)4 * 1024 * 1024; // [B,H,S,64] fp32

    char* ws = (char*)d_ws;
    const size_t MB = 1u << 20;
    unsigned short* query_bf = (unsigned short*)(ws + 0 * MB);   // 8 MB
    unsigned short* value_bf = (unsigned short*)(ws + 8 * MB);   // 8 MB
    unsigned short* Wq_t     = (unsigned short*)(ws + 16 * MB);  // 2 MB
    unsigned short* Wk_t     = (unsigned short*)(ws + 18 * MB);  // 2 MB
    unsigned short* Wv_t     = (unsigned short*)(ws + 20 * MB);  // 2 MB
    unsigned short* Wo_t     = (unsigned short*)(ws + 22 * MB);  // 2 MB
    unsigned short* q_bf     = (unsigned short*)(ws + 24 * MB);  // 8 MB
    unsigned short* k_bf     = (unsigned short*)(ws + 32 * MB);  // 8 MB
    unsigned short* v_bf     = (unsigned short*)(ws + 40 * MB);  // 8 MB
    unsigned short* v_t      = (unsigned short*)(ws + 48 * MB);  // 8 MB
    unsigned short* attn_bf  = (unsigned short*)(ws + 56 * MB);  // 8 MB

    const int n4 = BATCH * SEQ * DMODEL;                   // 4 M elements
    const int nW = 1024 * 1024;

    // Stage bf16 copies / transposed weights
    cvt_f32_bf16<<<n4 / 256, 256, 0, stream>>>(query, query_bf, n4);
    cvt_f32_bf16<<<n4 / 256, 256, 0, stream>>>(value, value_bf, n4);
    cvt_w_transpose<<<nW / 256, 256, 0, stream>>>(Wq, Wq_t);
    cvt_w_transpose<<<nW / 256, 256, 0, stream>>>(Wk, Wk_t);
    cvt_w_transpose<<<nW / 256, 256, 0, stream>>>(Wv, Wv_t);
    cvt_w_transpose<<<nW / 256, 256, 0, stream>>>(Wo, Wo_t);

    // Projections: 4096x1024x1024, bf16 outputs. NOTE: k uses query (reference bug).
    gemm_bf16_wmma<1><<<512, 256, 0, stream>>>(query_bf, Wq_t, bq, nullptr, q_bf, 4096, 1024);
    gemm_bf16_wmma<1><<<512, 256, 0, stream>>>(query_bf, Wk_t, bk, nullptr, k_bf, 4096, 1024);
    gemm_bf16_wmma<1><<<512, 256, 0, stream>>>(value_bf, Wv_t, bv, nullptr, v_bf, 4096, 1024);

    // Per-head V transpose for contiguous P@V B-fragments
    transpose_v<<<n4 / 256, 256, 0, stream>>>(v_bf, v_t);

    // Flash attention (4096 waves, 8 waves/block)
    attn_wmma<<<512, 256, 0, stream>>>(q_bf, k_bf, v_t, out_h, attn_bf);

    // Output projection to fp32
    gemm_bf16_wmma<0><<<512, 256, 0, stream>>>(attn_bf, Wo_t, bo, out_proj, nullptr, 4096, 1024);
}
